// ETM_7189775254159
// MI455X (gfx1250) — compile-verified, matrix operand order checked by reading, not run on hardware
//
#include <hip/hip_runtime.h>
#include <hip/hip_bf16.h>
#include <math.h>

// ---------------- problem constants ----------------
#define D_T   12
#define D_K   50
#define D_V   5000
#define D_B   256
#define D_R   300
#define D_HT  800
#define D_HE  256
#define LOG_DELTA (-5.298317366548036f)   // log(0.005)

// padded: K multiple of 32; A rows multiple of 32; Wt rows multiple of 64
#define R_P   320     // K: 300 -> 320
#define V_P   5024    // K: 5000 -> 5024
#define VK_P  5056    // K: 5050 -> 5056
#define KK_P  64      // K: 50 -> 64
#define TK_P  608     // A rows: 600 -> 608 (32-mult)
#define RW_V  5056    // Wt rows: 5000 -> 5056 (64-mult)
#define RW_HT 832     // Wt rows: 800 -> 832 (64-mult)

typedef __attribute__((ext_vector_type(16))) _Float16 v16h;
typedef __attribute__((ext_vector_type(8)))  _Float16 h8;
typedef __attribute__((ext_vector_type(8)))  float    v8f;

// ---------------- helpers ----------------
__device__ __forceinline__ unsigned pcg_hash(unsigned v) {
  unsigned s = v * 747796405u + 2891336453u;
  unsigned w = ((s >> ((s >> 28u) + 4u)) ^ s) * 277803737u;
  return (w >> 22u) ^ w;
}
__device__ __forceinline__ float rng_normal(unsigned idx, unsigned salt) {
  unsigned r1 = pcg_hash(idx ^ salt);
  unsigned r2 = pcg_hash(r1 + 0x9e3779b9u);
  float u1 = (float)(r1 >> 8) * (1.0f / 16777216.0f) + 1.0e-7f;
  float u2 = (float)(r2 >> 8) * (1.0f / 16777216.0f);
  return sqrtf(-2.0f * logf(u1)) * cosf(6.28318530718f * u2);
}
__device__ __forceinline__ float sigmoidf_(float x) { return 1.0f / (1.0f + expf(-x)); }
__device__ __forceinline__ float clip100(float x)   { return fminf(fmaxf(x, -100.f), 100.f); }

// one lane's 16-half fragment = two aligned 16B runs (ISA 7.12.2 16-bit layout)
__device__ __forceinline__ v16h ldfrag(const _Float16* __restrict__ p) {
  h8 lo = *(const h8*)(p);
  h8 hi = *(const h8*)(p + 16);
  return __builtin_shufflevector(lo, hi, 0,1,2,3,4,5,6,7,8,9,10,11,12,13,14,15);
}
#define WMMA(a, b, c) __builtin_amdgcn_wmma_f32_16x16x32_f16(false, (a), false, (b), (short)0, (c), false, false)

// ---------------- WMMA GEMM, 32x64 macro-tile per wave (2x4 subtiles) ----------------
// C[M,N] = act(A[M,Kpad] * Wt[N,Kpad]^T + bias); f16 operands, f32 accum.
// A rows padded to ceil32(M), Wt rows padded to ceil64(N); K-pads zero.
// FMODE 0: store C (f32) and/or C16 (f16)
// FMODE 1: atomicAdd(acc, fscale * sum(C .* Ff[ldf]))            (adjacency fuse)
// FMODE 2: atomicAdd(acc, fscale * sum(-log(C+1e-6) .* Fh[ldf])) (NLL fuse)
// ACCUM  : C += result
template <int RELU, int FMODE, int ACCUM>
__global__ __launch_bounds__(256)
void gemm16_tpl(const _Float16* __restrict__ A, const _Float16* __restrict__ Wt,
                const float* __restrict__ bias,
                float* __restrict__ C, _Float16* __restrict__ C16,
                const float* __restrict__ Ff, const _Float16* __restrict__ Fh,
                float* __restrict__ acc,
                int M, int N, int Kpad, int ldf, float fscale)
{
  const int lane = threadIdx.x & 31;
  const int wid  = threadIdx.x >> 5;
  const int tiles_n = (N + 63) >> 6;
  const int tiles_m = (M + 31) >> 5;
  long tile = (long)blockIdx.x * 8 + wid;
  if (tile >= (long)tiles_m * tiles_n) return;
  const int m0 = ((int)(tile / tiles_n)) << 5;
  const int n0 = ((int)(tile % tiles_n)) << 6;
  const int basek = (lane & 16) ? 8 : 0;
  const int lr = lane & 15;
  const _Float16* ap0 = A + (long)(m0 + lr) * Kpad + basek;
  const _Float16* ap1 = ap0 + (long)16 * Kpad;
  const _Float16* bp0 = Wt + (long)(n0 + lr) * Kpad + basek;
  const _Float16* bp1 = bp0 + (long)16 * Kpad;
  const _Float16* bp2 = bp0 + (long)32 * Kpad;
  const _Float16* bp3 = bp0 + (long)48 * Kpad;

  v8f cc[2][4] = {};
  for (int k0 = 0; k0 < Kpad; k0 += 32) {
    __builtin_prefetch((const void*)(ap0 + k0 + 512), 0, 1);
    __builtin_prefetch((const void*)(bp0 + k0 + 512), 0, 1);
    v16h a0 = ldfrag(ap0 + k0);
    v16h a1 = ldfrag(ap1 + k0);
    v16h b0 = ldfrag(bp0 + k0);
    v16h b1 = ldfrag(bp1 + k0);
    v16h b2 = ldfrag(bp2 + k0);
    v16h b3 = ldfrag(bp3 + k0);
    cc[0][0] = WMMA(a0, b0, cc[0][0]);
    cc[0][1] = WMMA(a0, b1, cc[0][1]);
    cc[0][2] = WMMA(a0, b2, cc[0][2]);
    cc[0][3] = WMMA(a0, b3, cc[0][3]);
    cc[1][0] = WMMA(a1, b0, cc[1][0]);
    cc[1][1] = WMMA(a1, b1, cc[1][1]);
    cc[1][2] = WMMA(a1, b2, cc[1][2]);
    cc[1][3] = WMMA(a1, b3, cc[1][3]);
  }

  const int mbo = (lane & 16) ? 8 : 0;
  float fsum = 0.0f;
#pragma unroll
  for (int i = 0; i < 2; ++i) {
#pragma unroll
    for (int j = 0; j < 4; ++j) {
      const int cn = n0 + j * 16 + lr;
      if (cn >= N) continue;
      const float bval = bias ? bias[cn] : 0.0f;
#pragma unroll
      for (int r = 0; r < 8; ++r) {
        const int cm = m0 + i * 16 + mbo + r;
        if (cm >= M) continue;
        float v = cc[i][j][r] + bval;
        if (ACCUM && C) v += C[(long)cm * N + cn];
        if (RELU) v = v > 0.0f ? v : 0.0f;
        if (FMODE == 1) {
          fsum += v * Ff[(long)cm * ldf + cn];
        } else if (FMODE == 2) {
          fsum += -logf(v + 1e-6f) * (float)Fh[(long)cm * ldf + cn];
        } else {
          if (C)   C[(long)cm * N + cn]   = v;
          if (C16) C16[(long)cm * N + cn] = (_Float16)v;
        }
      }
    }
  }
  if (FMODE) {
    for (int o = 16; o > 0; o >>= 1) fsum += __shfl_down(fsum, o, 32);
    if (lane == 0) atomicAdd(acc, fsum * fscale);
  }
}

template <int RELU, int FMODE, int ACCUM>
static void gemm(hipStream_t st, const _Float16* A, const _Float16* Wt, const float* bias,
                 float* C, _Float16* C16, const float* Ff, const _Float16* Fh, float* acc,
                 int M, int N, int Kpad, int ldf, float fscale) {
  long tiles = (long)((M + 31) / 32) * ((N + 63) / 64);
  int blocks = (int)((tiles + 7) / 8);
  gemm16_tpl<RELU, FMODE, ACCUM><<<blocks, 256, 0, st>>>(A, Wt, bias, C, C16, Ff, Fh, acc,
                                                         M, N, Kpad, ldf, fscale);
}

// ---------------- small kernels ----------------
__global__ void zero_f32_kernel(float* p, long n) {
  long i = (long)blockIdx.x * blockDim.x + threadIdx.x;
  if (i < n) p[i] = 0.0f;
}
__global__ void zero_f16_kernel(_Float16* p, long n) {
  long i = (long)blockIdx.x * blockDim.x + threadIdx.x;
  if (i < n) p[i] = (_Float16)0.0f;
}
// src f32 [rows, Ks] -> dst f16 [rows, Kpad], zero-filled K padding
__global__ void cvtpad_kernel(const float* __restrict__ s, _Float16* __restrict__ d,
                              int rows, int Ks, int Kpad) {
  long i = (long)blockIdx.x * blockDim.x + threadIdx.x;
  if (i >= (long)rows * Kpad) return;
  int k = (int)(i % Kpad);
  long r = i / Kpad;
  d[i] = (k < Ks) ? (_Float16)s[r * Ks + k] : (_Float16)0.0f;
}
// bow_t [B,T,V] f32 -> [T*B, V_P] f16 (zero pad)
__global__ void bowT_kernel(const float* __restrict__ bow, _Float16* __restrict__ dst) {
  const long n = (long)D_T * D_B * V_P;
  long i = (long)blockIdx.x * blockDim.x + threadIdx.x;
  if (i >= n) return;
  int v = (int)(i % V_P);
  long row = i / V_P;                    // t*B + b
  int b = (int)(row % D_B), t = (int)(row / D_B);
  dst[i] = (v < D_V) ? (_Float16)bow[((long)b * D_T + t) * D_V + v] : (_Float16)0.0f;
}
// adj [V,V] f32 -> adjT f16 [RW_V rows, V_P] with dst[n][k] = adj[k][n] (LDS tiled transpose)
__global__ __launch_bounds__(256)
void adjT_kernel(const float* __restrict__ adj, _Float16* __restrict__ dst) {
  __shared__ float tile[16][17];
  const int tx = threadIdx.x & 15, ty = threadIdx.x >> 4;
  const int k0 = blockIdx.x * 16;        // dst col / src row
  const int n0 = blockIdx.y * 16;        // dst row / src col
  const int sk = k0 + ty, sn = n0 + tx;
  tile[ty][tx] = (sk < D_V && sn < D_V) ? adj[(long)sk * D_V + sn] : 0.0f;
  __syncthreads();
  const int dn = n0 + ty, dk = k0 + tx;
  if (dn < D_V && dk < V_P)
    dst[(long)dn * V_P + dk] = (dk < D_V) ? (_Float16)tile[tx][ty] : (_Float16)0.0f;
}
// alphas[t,k,r] = mu_q_alpha[k,t,r] + eps*exp(0.5*ls); writes f32 [600,300] + padded f16 [608,320]
__global__ void alpha_sample_kernel(const float* __restrict__ mu, const float* __restrict__ ls,
                                    float* __restrict__ alphas, _Float16* __restrict__ alphas16) {
  const long n = (long)D_T * D_K * R_P;
  long i = (long)blockIdx.x * blockDim.x + threadIdx.x;
  if (i >= n) return;
  int r = (int)(i % R_P);
  long row = i / R_P;                    // t*K + k
  if (r >= D_R) { alphas16[i] = (_Float16)0.0f; return; }
  int k = (int)(row % D_K), t = (int)(row / D_K);
  long src = ((long)k * D_T + t) * D_R + r;
  float a = mu[src] + rng_normal((unsigned)(row * D_R + r), 0xA1FAu) * expf(0.5f * ls[src]);
  alphas[row * D_R + r] = a;
  alphas16[i] = (_Float16)a;
}
__global__ __launch_bounds__(256)
void kl_alpha_kernel(const float* __restrict__ mu, const float* __restrict__ ls,
                     const float* __restrict__ alphas, float* __restrict__ out) {
  __shared__ float red[256];
  const long n = (long)D_T * D_K * D_R;
  long i = (long)blockIdx.x * 256 + threadIdx.x;
  float kl = 0.0f;
  if (i < n) {
    int r = (int)(i % D_R);
    long tk = i / D_R;
    int k = (int)(tk % D_K), t = (int)(tk / D_K);
    long src = ((long)k * D_T + t) * D_R + r;
    float m = mu[src];
    float q = clip100(ls[src]);
    float pmu = (t == 0) ? 0.0f : alphas[i - (long)D_K * D_R];
    float pls = (t == 0) ? 0.0f : LOG_DELTA;
    float d = m - pmu;
    kl = (expf(q) + d * d) / (expf(pls) + 1e-6f) - 1.0f + pls - q;
  }
  red[threadIdx.x] = kl; __syncthreads();
  for (int s = 128; s > 0; s >>= 1) {
    if ((int)threadIdx.x < s) red[threadIdx.x] += red[threadIdx.x + s];
    __syncthreads();
  }
  if (threadIdx.x == 0) atomicAdd(out, 0.5f * red[0] / (float)(D_T * D_K));
}
// row softmax over V; writes beta f32 [600,5000], beta16 [608,V_P], betaT16 [T][RW_V][KK_P]
__global__ __launch_bounds__(256)
void softmax_beta_kernel(const float* __restrict__ logits, float* __restrict__ beta,
                         _Float16* __restrict__ beta16, _Float16* __restrict__ betaT16) {
  const int row = blockIdx.x;            // t*K + k
  const int t = row / D_K, k = row % D_K;
  const float* lr = logits + (long)row * D_V;
  __shared__ float red[256];
  float mx = -3.4e38f;
  for (int v = threadIdx.x; v < D_V; v += 256) mx = fmaxf(mx, lr[v]);
  red[threadIdx.x] = mx; __syncthreads();
  for (int s = 128; s > 0; s >>= 1) {
    if ((int)threadIdx.x < s) red[threadIdx.x] = fmaxf(red[threadIdx.x], red[threadIdx.x + s]);
    __syncthreads();
  }
  mx = red[0]; __syncthreads();
  float sum = 0.0f;
  for (int v = threadIdx.x; v < D_V; v += 256) sum += expf(lr[v] - mx);
  red[threadIdx.x] = sum; __syncthreads();
  for (int s = 128; s > 0; s >>= 1) {
    if ((int)threadIdx.x < s) red[threadIdx.x] += red[threadIdx.x + s];
    __syncthreads();
  }
  const float inv = 1.0f / red[0];
  for (int v = threadIdx.x; v < V_P; v += 256) {
    if (v < D_V) {
      float bv = expf(lr[v] - mx) * inv;
      beta[(long)row * D_V + v]    = bv;
      beta16[(long)row * V_P + v]  = (_Float16)bv;
      betaT16[((long)t * RW_V + v) * KK_P + k] = (_Float16)bv;
    } else {
      beta16[(long)row * V_P + v] = (_Float16)0.0f;
    }
  }
}
__global__ void lstm_cell_kernel(const float* __restrict__ g, float* __restrict__ c,
                                 float* __restrict__ hout, _Float16* __restrict__ h16) {
  int i = blockIdx.x * blockDim.x + threadIdx.x;
  if (i >= D_B * D_HE) return;
  int b = i / D_HE, j = i % D_HE;
  const float* gb = g + (long)b * 4 * D_HE;
  float ig = sigmoidf_(gb[j]);
  float fg = sigmoidf_(gb[D_HE + j]);
  float gg = tanhf(gb[2 * D_HE + j]);
  float og = sigmoidf_(gb[3 * D_HE + j]);
  float cn = fg * c[i] + ig * gg;
  float h  = og * tanhf(cn);
  c[i] = cn; hout[i] = h; h16[i] = (_Float16)h;
}
__global__ __launch_bounds__(256)
void eta_step_kernel(const float* __restrict__ o, const float* __restrict__ prev,
                     const float* __restrict__ mueW, const float* __restrict__ mueB,
                     const float* __restrict__ lseW, const float* __restrict__ lseB,
                     float* __restrict__ eta_out, float* __restrict__ kld,
                     float pls, int t) {
  __shared__ float red[256];
  int i = blockIdx.x * 256 + threadIdx.x;
  float kl = 0.0f;
  if (i < D_B * D_K) {
    int b = i / D_K, k = i % D_K;
    const float* wm = mueW + (long)k * (D_HE + D_K);
    const float* wl = lseW + (long)k * (D_HE + D_K);
    const float* ob = o + (long)b * D_HE;
    const float* pb = prev + (long)b * D_K;
    float mu = mueB[k], lsv = lseB[k];
    for (int j = 0; j < D_HE; ++j) { float x = ob[j]; mu += wm[j] * x; lsv += wl[j] * x; }
    for (int j = 0; j < D_K;  ++j) { float x = pb[j]; mu += wm[D_HE + j] * x; lsv += wl[D_HE + j] * x; }
    float eps = rng_normal((unsigned)(t * D_B * D_K + i), 0x5E7Au);
    eta_out[i] = mu + eps * expf(0.5f * lsv);
    float q = clip100(lsv);
    float d = mu - pb[k];
    kl = (expf(q) + d * d) / (expf(pls) + 1e-6f) - 1.0f + pls - q;
  }
  red[threadIdx.x] = kl; __syncthreads();
  for (int s = 128; s > 0; s >>= 1) {
    if ((int)threadIdx.x < s) red[threadIdx.x] += red[threadIdx.x + s];
    __syncthreads();
  }
  if (threadIdx.x == 0) atomicAdd(kld, 0.5f * red[0] / (float)D_B);
}
// x16[t*B+b, VK_P] = [ bows[b,:V], etas[t,b,:K], zeros ]
__global__ void buildx_kernel(const float* __restrict__ bows, const float* __restrict__ etas,
                              _Float16* __restrict__ x) {
  const long n = (long)D_T * D_B * VK_P;
  long i = (long)blockIdx.x * blockDim.x + threadIdx.x;
  if (i >= n) return;
  int c = (int)(i % VK_P);
  long row = i / VK_P;
  int b = (int)(row % D_B);
  float v = 0.0f;
  if (c < D_V)            v = bows[(long)b * D_V + c];
  else if (c < D_V + D_K) v = etas[row * D_K + (c - D_V)];
  x[i] = (_Float16)v;
}
__global__ __launch_bounds__(64)
void theta_sample_kernel(const float* __restrict__ mu_t, const float* __restrict__ ls_t,
                         const float* __restrict__ etas, _Float16* __restrict__ theta16,
                         float* __restrict__ kld_out) {
  const int row = blockIdx.x;            // t*B + b
  const int k = threadIdx.x;
  __shared__ float red[64];
  float z = -3.4e38f, mu = 0.0f, ls = 0.0f, kl = 0.0f;
  if (k < D_K) {
    mu = mu_t[(long)row * D_K + k];
    ls = ls_t[(long)row * D_K + k];
    float eps = rng_normal((unsigned)(row * D_K + k), 0x7E7Au);
    z = mu + eps * expf(0.5f * ls);
    float q = clip100(ls);
    float e = etas[(long)row * D_K + k];
    kl = (expf(q) + (mu - e) * (mu - e)) / (1.0f + 1e-6f) - 1.0f - q;
  }
  red[k] = z; __syncthreads();
  for (int s = 32; s > 0; s >>= 1) { if (k < s) red[k] = fmaxf(red[k], red[k + s]); __syncthreads(); }
  float mx = red[0]; __syncthreads();
  float ez = (k < D_K) ? expf(z - mx) : 0.0f;
  red[k] = ez; __syncthreads();
  for (int s = 32; s > 0; s >>= 1) { if (k < s) red[k] += red[k + s]; __syncthreads(); }
  float sum = red[0]; __syncthreads();
  theta16[(long)row * KK_P + k] = (k < D_K) ? (_Float16)(ez / sum) : (_Float16)0.0f;
  red[k] = kl; __syncthreads();
  for (int s = 32; s > 0; s >>= 1) { if (k < s) red[k] += red[k + s]; __syncthreads(); }
  if (k == 0) atomicAdd(kld_out, 0.5f * red[0] / (float)(D_T * D_B));
}

// ---------------- host side ----------------
extern "C" void kernel_launch(void* const* d_in, const int* in_sizes, int n_in,
                              void* d_out, int out_size, void* d_ws, size_t ws_size,
                              hipStream_t stream) {
  (void)in_sizes; (void)n_in; (void)out_size; (void)ws_size;
  const float* bows   = (const float*)d_in[0];
  const float* bow_t  = (const float*)d_in[1];
  const float* adj    = (const float*)d_in[2];
  const float* mu_qa  = (const float*)d_in[3];
  const float* ls_qa  = (const float*)d_in[4];
  const float* rho_W  = (const float*)d_in[5];
  const float* rho_b  = (const float*)d_in[6];
  const float* qt_W1  = (const float*)d_in[7];
  const float* qt_b1  = (const float*)d_in[8];
  const float* qt_W2  = (const float*)d_in[9];
  const float* qt_b2  = (const float*)d_in[10];
  const float* mut_W  = (const float*)d_in[11];
  const float* mut_b  = (const float*)d_in[12];
  const float* lst_W  = (const float*)d_in[13];
  const float* lst_b  = (const float*)d_in[14];
  const float* qe_W   = (const float*)d_in[15];
  const float* qe_b   = (const float*)d_in[16];
  const float* Wih    = (const float*)d_in[17];
  const float* Whh    = (const float*)d_in[18];
  const float* bih    = (const float*)d_in[19];
  const float* bhh    = (const float*)d_in[20];
  const float* mue_W  = (const float*)d_in[21];
  const float* mue_b  = (const float*)d_in[22];
  const float* lse_W  = (const float*)d_in[23];
  const float* lse_b  = (const float*)d_in[24];
  float* out = (float*)d_out;   // [nll, kld_theta, kld_alpha, kld_eta, cond]

  size_t off = 0;
  char* base = (char*)d_ws;
  auto alloc = [&](size_t elems, size_t esz) -> void* {
    void* p = base + off;
    off += ((elems * esz + 255) / 256) * 256;
    return p;
  };
  float*    alphas   = (float*)   alloc((size_t)D_T * D_K * D_R, 4);
  _Float16* alphas16 = (_Float16*)alloc((size_t)TK_P * R_P, 2);
  _Float16* rhoW16   = (_Float16*)alloc((size_t)RW_V * R_P, 2);
  float*    logits   = (float*)   alloc((size_t)D_T * D_K * D_V, 4);
  float*    beta     = (float*)   alloc((size_t)D_T * D_K * D_V, 4);
  _Float16* beta16   = (_Float16*)alloc((size_t)TK_P * V_P, 2);
  _Float16* betaT16  = (_Float16*)alloc((size_t)D_T * RW_V * KK_P, 2);
  _Float16* adjT16   = (_Float16*)alloc((size_t)RW_V * V_P, 2);
  _Float16* bowT16   = (_Float16*)alloc((size_t)D_T * D_B * V_P, 2);
  _Float16* qeW16    = (_Float16*)alloc((size_t)D_HE * V_P, 2);
  _Float16* inp16    = (_Float16*)alloc((size_t)D_T * D_B * D_HE, 2);
  _Float16* Wih16    = (_Float16*)alloc((size_t)4 * D_HE * D_HE, 2);
  _Float16* Whh16    = (_Float16*)alloc((size_t)4 * D_HE * D_HE, 2);
  _Float16* h16      = (_Float16*)alloc((size_t)D_B * D_HE, 2);
  float*    cbuf     = (float*)   alloc((size_t)D_B * D_HE, 4);
  float*    gbuf     = (float*)   alloc((size_t)D_B * 4 * D_HE, 4);
  float*    lstm_out = (float*)   alloc((size_t)D_T * D_B * D_HE, 4);
  float*    etas     = (float*)   alloc((size_t)D_T * D_B * D_K, 4);
  float*    eta_zero = (float*)   alloc((size_t)D_B * D_K, 4);
  _Float16* x16      = (_Float16*)alloc((size_t)D_T * D_B * VK_P, 2);
  _Float16* qtW1_16  = (_Float16*)alloc((size_t)RW_HT * VK_P, 2);
  _Float16* qtW2_16  = (_Float16*)alloc((size_t)RW_HT * D_HT, 2);
  _Float16* mutW16   = (_Float16*)alloc((size_t)KK_P * D_HT, 2);
  _Float16* lstW16   = (_Float16*)alloc((size_t)KK_P * D_HT, 2);
  _Float16* h1_16    = (_Float16*)alloc((size_t)D_T * D_B * D_HT, 2);
  _Float16* h2_16    = (_Float16*)alloc((size_t)D_T * D_B * D_HT, 2);
  float*    mu_t     = (float*)   alloc((size_t)D_T * D_B * D_K, 4);
  float*    ls_t     = (float*)   alloc((size_t)D_T * D_B * D_K, 4);
  _Float16* theta16  = (_Float16*)alloc((size_t)D_T * D_B * KK_P, 2);

  const int TPB = 256;
  auto blk = [](long n) { return (int)((n + 255) / 256); };

  // -------- init --------
  zero_f32_kernel<<<1, 32, 0, stream>>>(out, 5);
  zero_f32_kernel<<<blk((long)D_B * D_K), TPB, 0, stream>>>(eta_zero, (long)D_B * D_K);
  zero_f32_kernel<<<blk((long)D_B * D_HE), TPB, 0, stream>>>(cbuf, (long)D_B * D_HE);
  zero_f16_kernel<<<blk((long)D_B * D_HE), TPB, 0, stream>>>(h16, (long)D_B * D_HE);
  zero_f16_kernel<<<blk((long)D_T * RW_V * KK_P), TPB, 0, stream>>>(betaT16, (long)D_T * RW_V * KK_P);

  // -------- weight conversions (pad K to mult-of-32, zero-filled) --------
  cvtpad_kernel<<<blk((long)D_V * R_P), TPB, 0, stream>>>(rho_W, rhoW16, D_V, D_R, R_P);
  cvtpad_kernel<<<blk((long)D_HE * V_P), TPB, 0, stream>>>(qe_W, qeW16, D_HE, D_V, V_P);
  cvtpad_kernel<<<blk((long)4 * D_HE * D_HE), TPB, 0, stream>>>(Wih, Wih16, 4 * D_HE, D_HE, D_HE);
  cvtpad_kernel<<<blk((long)4 * D_HE * D_HE), TPB, 0, stream>>>(Whh, Whh16, 4 * D_HE, D_HE, D_HE);
  cvtpad_kernel<<<blk((long)D_HT * VK_P), TPB, 0, stream>>>(qt_W1, qtW1_16, D_HT, D_V + D_K, VK_P);
  cvtpad_kernel<<<blk((long)D_HT * D_HT), TPB, 0, stream>>>(qt_W2, qtW2_16, D_HT, D_HT, D_HT);
  cvtpad_kernel<<<blk((long)D_K * D_HT), TPB, 0, stream>>>(mut_W, mutW16, D_K, D_HT, D_HT);
  cvtpad_kernel<<<blk((long)D_K * D_HT), TPB, 0, stream>>>(lst_W, lstW16, D_K, D_HT, D_HT);
  bowT_kernel<<<blk((long)D_T * D_B * V_P), TPB, 0, stream>>>(bow_t, bowT16);
  {
    dim3 g(V_P / 16, (D_V + 15) / 16);
    adjT_kernel<<<g, 256, 0, stream>>>(adj, adjT16);
  }

  // -------- alpha: sample + KL --------
  alpha_sample_kernel<<<blk((long)D_T * D_K * R_P), TPB, 0, stream>>>(mu_qa, ls_qa, alphas, alphas16);
  kl_alpha_kernel<<<blk((long)D_T * D_K * D_R), TPB, 0, stream>>>(mu_qa, ls_qa, alphas, &out[2]);

  // -------- beta: logits = alphas @ rho_W^T + rho_b ; softmax over V --------
  gemm<0, 0, 0>(stream, alphas16, rhoW16, rho_b, logits, nullptr, nullptr, nullptr, nullptr,
                D_T * D_K, D_V, R_P, 0, 0.0f);
  softmax_beta_kernel<<<D_T * D_K, 256, 0, stream>>>(logits, beta, beta16, betaT16);

  // -------- eta path: inp = bow^T @ qe_W^T + qe_b -> [T,B,HE] f16 --------
  gemm<0, 0, 0>(stream, bowT16, qeW16, qe_b, nullptr, inp16, nullptr, nullptr, nullptr,
                D_T * D_B, D_HE, V_P, 0, 0.0f);

  // -------- LSTM over T --------
  for (int t = 0; t < D_T; ++t) {
    gemm<0, 0, 0>(stream, inp16 + (long)t * D_B * D_HE, Wih16, bih, gbuf, nullptr,
                  nullptr, nullptr, nullptr, D_B, 4 * D_HE, D_HE, 0, 0.0f);
    gemm<0, 0, 1>(stream, h16, Whh16, bhh, gbuf, nullptr,
                  nullptr, nullptr, nullptr, D_B, 4 * D_HE, D_HE, 0, 0.0f);
    lstm_cell_kernel<<<blk((long)D_B * D_HE), TPB, 0, stream>>>(
        gbuf, cbuf, lstm_out + (long)t * D_B * D_HE, h16);
  }

  // -------- eta recurrence + KL --------
  for (int t = 0; t < D_T; ++t) {
    const float* prev = (t == 0) ? eta_zero : etas + (long)(t - 1) * D_B * D_K;
    float pls = (t == 0) ? 0.0f : LOG_DELTA;
    eta_step_kernel<<<blk((long)D_B * D_K), TPB, 0, stream>>>(
        lstm_out + (long)t * D_B * D_HE, prev, mue_W, mue_b, lse_W, lse_b,
        etas + (long)t * D_B * D_K, &out[3], pls, t);
  }

  // -------- theta encoder MLP --------
  buildx_kernel<<<blk((long)D_T * D_B * VK_P), TPB, 0, stream>>>(bows, etas, x16);
  gemm<1, 0, 0>(stream, x16, qtW1_16, qt_b1, nullptr, h1_16, nullptr, nullptr, nullptr,
                D_T * D_B, D_HT, VK_P, 0, 0.0f);
  gemm<1, 0, 0>(stream, h1_16, qtW2_16, qt_b2, nullptr, h2_16, nullptr, nullptr, nullptr,
                D_T * D_B, D_HT, D_HT, 0, 0.0f);
  gemm<0, 0, 0>(stream, h2_16, mutW16, mut_b, mu_t, nullptr, nullptr, nullptr, nullptr,
                D_T * D_B, D_K, D_HT, 0, 0.0f);
  gemm<0, 0, 0>(stream, h2_16, lstW16, lst_b, ls_t, nullptr, nullptr, nullptr, nullptr,
                D_T * D_B, D_K, D_HT, 0, 0.0f);
  theta_sample_kernel<<<D_T * D_B, 64, 0, stream>>>(mu_t, ls_t, etas, theta16, &out[1]);

  // -------- NLL: lik = theta @ beta per t, fused -10/B * sum(log(lik+eps)*bow) --------
  for (int t = 0; t < D_T; ++t) {
    gemm<0, 2, 0>(stream, theta16 + (long)t * D_B * KK_P, betaT16 + (long)t * RW_V * KK_P,
                  nullptr, nullptr, nullptr, nullptr, bowT16 + (long)t * D_B * V_P, &out[0],
                  D_B, D_V, KK_P, V_P, 10.0f / (float)D_B);
  }

  // -------- adjacency loss: tmp = beta @ adj, fused -lambda * sum(tmp .* beta) --------
  gemm<0, 1, 0>(stream, beta16, adjT16, nullptr, nullptr, nullptr, beta, nullptr, &out[4],
                D_T * D_K, D_V, V_P, D_V, -0.01f);
}
